// EfficientAttention_51702816309817
// MI455X (gfx1250) — matrix operand order repaired and checked by reading, not compiled, not run on hardware
//
#include <hip/hip_runtime.h>

typedef __attribute__((ext_vector_type(16))) __bf16   v16bf;
typedef __attribute__((ext_vector_type(8)))  float    v8f;
typedef __attribute__((ext_vector_type(4)))  unsigned v4u;
typedef __attribute__((ext_vector_type(8)))  int      v8i;
typedef __attribute__((ext_vector_type(4)))  int      v4i;

#define CDIM  256
#define HWDIM 4096
#define NBAT  16

// ---- helpers -------------------------------------------------------------

__device__ __forceinline__ __bf16 f2b(float f) {
  unsigned u = __builtin_bit_cast(unsigned, f);
  unsigned short s = (unsigned short)((u + 0x7FFFu + ((u >> 16) & 1u)) >> 16);
  return __builtin_bit_cast(__bf16, s);
}

// K offset of the low element of VGPR j for a 16-bit A fragment (16x32),
// per CDNA5 ISA 7.12.2: V0..3 = K{0..7}/K{8..15}, V4..7 = K{16..31}.
__device__ __forceinline__ int a_kb(int j, int half) {
  return (j < 4) ? (half * 8 + 2 * j) : (16 + half * 8 + 2 * (j - 4));
}

union AF { v16bf v; unsigned u[8]; unsigned short s[16]; };
union CF { v8f   v; float    f[8]; };

// ---- fused 256x256xL GEMM (bias, optional residual, optional bf16 B) -----
// Out[n][m][l] = sum_k W[m][k] * B[n][k][l] + bias[m] (+ Res[n][m][l])
// Block tile: 128(M) x 128(N); 8 waves (4x2), each a 32x64 sub-tile
// (2x4 WMMA frags -> 8 v_wmma per K-step per wave).
// bf16-B path stages its tile with the Tensor Data Mover (TENSORcnt).

template <bool B_IS_BF16, bool RESIDUAL>
__global__ void __launch_bounds__(256)
gemm256(const float* __restrict__ W, const float* __restrict__ bias,
        const void* __restrict__ Bsrc, const float* __restrict__ Res,
        float* __restrict__ Out)
{
  __shared__ __bf16 As[128 * 34];   // row-major [m][k], stride 34
  __shared__ __bf16 Bs[128 * 34];   // fp32 path: col-major [l][k] stride 34
                                    // bf16 path: row-major [k][128] (TDM tile)

  const int n  = blockIdx.z;
  const int m0 = blockIdx.y * 128;
  const int n0 = blockIdx.x * 128;
  const int t  = threadIdx.x;
  const int lane = t & 31, wave = t >> 5;
  const int waveM = wave >> 1, waveN = wave & 1;
  const int l15 = lane & 15, half = (lane >> 4) & 1;

  const float*          Bf = (const float*)Bsrc          + (size_t)n * CDIM * HWDIM;
  const unsigned short* Bb = (const unsigned short*)Bsrc + (size_t)n * CDIM * HWDIM;
  Out += (size_t)n * CDIM * HWDIM;
  const float* Resn = RESIDUAL ? (Res + (size_t)n * CDIM * HWDIM) : nullptr;

  CF acc[2][4];
  #pragma unroll
  for (int i = 0; i < 2; i++)
    #pragma unroll
    for (int j = 0; j < 4; j++)
      #pragma unroll
      for (int q = 0; q < 8; q++) acc[i][j].f[q] = 0.0f;

  for (int k0 = 0; k0 < CDIM; k0 += 32) {
    __syncthreads();

    // stage A tile (128x32 of W) as bf16
    #pragma unroll
    for (int i = t; i < 128 * 32; i += 256) {
      int r = i >> 5, c = i & 31;
      As[r * 34 + c] = f2b(W[(size_t)(m0 + r) * CDIM + k0 + c]);
    }

    if (B_IS_BF16) {
#if __has_builtin(__builtin_amdgcn_tensor_load_to_lds)
      // Tensor Data Mover: DMA the 32x128 bf16 tile (row-major) into LDS.
      if (wave == 0) {
        unsigned ldsoff = (unsigned)(size_t)&Bs[0];
        unsigned long long ga =
            (unsigned long long)(const void*)(Bb + (size_t)k0 * HWDIM + n0);
        v4u g0; v8i g1; v4i g2; v4i g3; v8i g4;
        g2[0] = g2[1] = g2[2] = g2[3] = 0;
        g3[0] = g3[1] = g3[2] = g3[3] = 0;
        #pragma unroll
        for (int q = 0; q < 8; q++) g4[q] = 0;
        // D# group0: count=1 | lds_addr | global_addr(57b) | type=2
        g0[0] = 1u;
        g0[1] = ldsoff;
        g0[2] = (unsigned)(ga & 0xFFFFFFFFull);
        g0[3] = (unsigned)((ga >> 32) & 0x01FFFFFFull) | (2u << 30);
        // D# group1: data_size=1 (2B); tensor 4096x256; tile 128x32; stride 4096
        g1[0] = 0x00010000;                               // wg_mask=0, data_size=2B
        g1[1] = (int)((HWDIM & 0xFFFFu) << 16);           // tensor_dim0[15:0]
        g1[2] = (int)((HWDIM >> 16) | ((CDIM & 0xFFFFu) << 16)); // dim0 hi | dim1 lo
        g1[3] = (int)((CDIM >> 16) | (128u << 16));       // dim1 hi | tile_dim0=128
        g1[4] = 32;                                       // tile_dim1=32
        g1[5] = HWDIM;                                    // tensor_dim0_stride lo
        g1[6] = 0;
        g1[7] = 0;
        __builtin_amdgcn_tensor_load_to_lds(g0, g1, g2, g3, g4, 0);
        __builtin_amdgcn_s_wait_tensorcnt(0);
      }
#else
      // fallback: manual row-major copy
      for (int i = t; i < 32 * 128; i += 256) {
        int k = i >> 7, col = i & 127;
        Bs[k * 128 + col] =
            __builtin_bit_cast(__bf16, Bb[(size_t)(k0 + k) * HWDIM + n0 + col]);
      }
#endif
    } else {
      // stage B tile (32x128 of fp32 activations), transposed to [col][k]
      #pragma unroll
      for (int i = t; i < 32 * 128; i += 256) {
        int k = i >> 7, col = i & 127;
        Bs[col * 34 + k] = f2b(Bf[(size_t)(k0 + k) * HWDIM + n0 + col]);
      }
      if (k0 + 32 < CDIM)
        __builtin_prefetch(&W[(size_t)m0 * CDIM + k0 + 32], 0, 1);
    }
    __syncthreads();

    AF a[2], b[4];
    #pragma unroll
    for (int i = 0; i < 2; i++) {
      int row = waveM * 32 + i * 16 + l15;
      #pragma unroll
      for (int j = 0; j < 8; j++)
        a[i].u[j] = *(const unsigned*)&As[row * 34 + a_kb(j, half)];
    }
    #pragma unroll
    for (int j2 = 0; j2 < 4; j2++) {
      int col = waveN * 64 + j2 * 16 + l15;
      if (B_IS_BF16) {
        // row-major tile: K-pairs span rows -> per-element LDS reads
        #pragma unroll
        for (int j = 0; j < 8; j++) {
          int kk = a_kb(j, half);
          b[j2].s[2 * j]     = __builtin_bit_cast(unsigned short, Bs[(kk)     * 128 + col]);
          b[j2].s[2 * j + 1] = __builtin_bit_cast(unsigned short, Bs[(kk + 1) * 128 + col]);
        }
      } else {
        #pragma unroll
        for (int j = 0; j < 8; j++)
          b[j2].u[j] = *(const unsigned*)&Bs[col * 34 + a_kb(j, half)];
      }
    }
    #pragma unroll
    for (int i = 0; i < 2; i++)
      #pragma unroll
      for (int j2 = 0; j2 < 4; j2++)
        acc[i][j2].v = __builtin_amdgcn_wmma_f32_16x16x32_bf16(
            false, a[i].v, false, b[j2].v, (short)0, acc[i][j2].v, false, false);
  }

  #pragma unroll
  for (int i = 0; i < 2; i++)
    #pragma unroll
    for (int j2 = 0; j2 < 4; j2++)
      #pragma unroll
      for (int j = 0; j < 8; j++) {
        int row = m0 + waveM * 32 + i * 16 + j + 8 * half;
        int col = n0 + waveN * 64 + j2 * 16 + l15;
        float vo = acc[i][j2].f[j] + bias[row];
        if (RESIDUAL) vo += Resn[(size_t)row * HWDIM + col];
        Out[(size_t)row * HWDIM + col] = vo;
      }
}

// ---- softmax over spatial dim (keys): one block per (n, channel) row -----

__global__ void __launch_bounds__(256)
softmax_rows(const float* __restrict__ In, __bf16* __restrict__ Ob)
{
  __shared__ float red[8];
  const int row = blockIdx.x;                 // n*256 + c
  const float* p = In + (size_t)row * HWDIM;
  const int t = threadIdx.x, lane = t & 31, wave = t >> 5;

  float vals[16];
  float m = -3.0e38f;
  #pragma unroll
  for (int i = 0; i < 16; i++) { vals[i] = p[t + i * 256]; m = fmaxf(m, vals[i]); }
  #pragma unroll
  for (int off = 16; off; off >>= 1) m = fmaxf(m, __shfl_xor(m, off, 32));
  if (lane == 0) red[wave] = m;
  __syncthreads();
  float bm = red[0];
  #pragma unroll
  for (int w = 1; w < 8; w++) bm = fmaxf(bm, red[w]);

  float s = 0.0f;
  #pragma unroll
  for (int i = 0; i < 16; i++) { vals[i] = __expf(vals[i] - bm); s += vals[i]; }
  #pragma unroll
  for (int off = 16; off; off >>= 1) s += __shfl_xor(s, off, 32);
  __syncthreads();
  if (lane == 0) red[wave] = s;
  __syncthreads();
  float bs = 0.0f;
  #pragma unroll
  for (int w = 0; w < 8; w++) bs += red[w];
  float inv = 1.0f / bs;

  __bf16* o = Ob + (size_t)row * HWDIM;
  #pragma unroll
  for (int i = 0; i < 16; i++) o[t + i * 256] = f2b(vals[i] * inv);
}

// ---- softmax over key-channels (queries): one thread per (n, h, l) -------

__global__ void __launch_bounds__(256)
softmax_chan(const float* __restrict__ Q, __bf16* __restrict__ Ob)
{
  const int tid  = blockIdx.x * 256 + threadIdx.x;
  const int l    = tid & (HWDIM - 1);
  const int rest = tid >> 12;
  const int h = rest & 7, n = rest >> 3;
  const size_t base = ((size_t)n * CDIM + h * 32) * HWDIM + l;
  const float* p = Q + base;

  float v[32];
  float m = -3.0e38f;
  #pragma unroll
  for (int k = 0; k < 32; k++) { v[k] = p[(size_t)k * HWDIM]; m = fmaxf(m, v[k]); }
  float s = 0.0f;
  #pragma unroll
  for (int k = 0; k < 32; k++) { v[k] = __expf(v[k] - m); s += v[k]; }
  float inv = 1.0f / s;
  __bf16* o = Ob + base;
  #pragma unroll
  for (int k = 0; k < 32; k++) o[(size_t)k * HWDIM] = f2b(v[k] * inv);
}

// ---- fp32 -> bf16 convert (values) ---------------------------------------

__global__ void __launch_bounds__(256)
cvt_bf16(const float* __restrict__ In, __bf16* __restrict__ Ob)
{
  const int i0 = (blockIdx.x * 256 + threadIdx.x) * 4;
  #pragma unroll
  for (int j = 0; j < 4; j++) Ob[i0 + j] = f2b(In[i0 + j]);
}

// ---- ctx = softmax(k) @ v^T : 32x32 per (n, head), K-dim = 4096 ----------
// 4 waves split L; partials reduced through LDS.

__global__ void __launch_bounds__(128)
ctx_kernel(const __bf16* __restrict__ kb, const __bf16* __restrict__ vb,
           float* __restrict__ ctx)
{
  __shared__ float part[4 * 32 * 32];
  const int nh = blockIdx.x;                            // n*8 + h
  const __bf16* K = kb + (size_t)nh * 32 * HWDIM;
  const __bf16* V = vb + (size_t)nh * 32 * HWDIM;
  const int t = threadIdx.x, lane = t & 31, wave = t >> 5;
  const int l15 = lane & 15, half = (lane >> 4) & 1;

  CF acc[2][2];
  #pragma unroll
  for (int i = 0; i < 2; i++)
    #pragma unroll
    for (int j = 0; j < 2; j++)
      #pragma unroll
      for (int q = 0; q < 8; q++) acc[i][j].f[q] = 0.0f;

  const int lbase = wave * 1024;
  for (int l0 = lbase; l0 < lbase + 1024; l0 += 32) {
    AF a[2], b[2];
    #pragma unroll
    for (int i = 0; i < 2; i++) {
      int r = i * 16 + l15;                              // key channel (M)
      #pragma unroll
      for (int j = 0; j < 8; j++)
        a[i].u[j] = *(const unsigned*)&K[(size_t)r * HWDIM + l0 + a_kb(j, half)];
    }
    #pragma unroll
    for (int j2 = 0; j2 < 2; j2++) {
      int vc = j2 * 16 + l15;                            // value channel (N)
      #pragma unroll
      for (int j = 0; j < 8; j++)
        b[j2].u[j] = *(const unsigned*)&V[(size_t)vc * HWDIM + l0 + a_kb(j, half)];
    }
    #pragma unroll
    for (int i = 0; i < 2; i++)
      #pragma unroll
      for (int j2 = 0; j2 < 2; j2++)
        acc[i][j2].v = __builtin_amdgcn_wmma_f32_16x16x32_bf16(
            false, a[i].v, false, b[j2].v, (short)0, acc[i][j2].v, false, false);
  }

  #pragma unroll
  for (int i = 0; i < 2; i++)
    #pragma unroll
    for (int j2 = 0; j2 < 2; j2++)
      #pragma unroll
      for (int j = 0; j < 8; j++) {
        int r = i * 16 + j + 8 * half;
        int c = j2 * 16 + l15;
        part[wave * 1024 + r * 32 + c] = acc[i][j2].f[j];
      }
  __syncthreads();
  float* out = ctx + (size_t)nh * 1024;
  for (int e = t; e < 1024; e += 128)
    out[e] = part[e] + part[1024 + e] + part[2048 + e] + part[3072 + e];
}

// ---- att = ctx^T @ softmax(q) : 32 x 4096 per (n, head), single K=32 -----

__global__ void __launch_bounds__(256)
att_kernel(const float* __restrict__ ctx, const __bf16* __restrict__ qb,
           __bf16* __restrict__ aggb)
{
  const int n = blockIdx.z, h = blockIdx.y;
  const int nh = n * 8 + h;
  const float* C = ctx + (size_t)nh * 1024;              // [k][v] 32x32
  const unsigned short* Qs = (const unsigned short*)(qb + (size_t)nh * 32 * HWDIM);
  __bf16* O = aggb + (size_t)nh * 32 * HWDIM;
  const int t = threadIdx.x, lane = t & 31, wave = t >> 5;
  const int l15 = lane & 15, half = (lane >> 4) & 1;
  const int col0 = blockIdx.x * 128 + wave * 16;

  AF a[2], b;
  #pragma unroll
  for (int i = 0; i < 2; i++) {
    int vrow = i * 16 + l15;                             // value channel (M)
    #pragma unroll
    for (int j = 0; j < 8; j++) {
      int k0 = a_kb(j, half);                            // key channel (K)
      a[i].s[2 * j]     = __builtin_bit_cast(unsigned short, f2b(C[(k0)     * 32 + vrow]));
      a[i].s[2 * j + 1] = __builtin_bit_cast(unsigned short, f2b(C[(k0 + 1) * 32 + vrow]));
    }
  }
  #pragma unroll
  for (int j = 0; j < 8; j++) {
    int k0 = a_kb(j, half);
    b.s[2 * j]     = Qs[(size_t)(k0)     * HWDIM + col0 + l15];
    b.s[2 * j + 1] = Qs[(size_t)(k0 + 1) * HWDIM + col0 + l15];
  }

  CF acc[2];
  #pragma unroll
  for (int i = 0; i < 2; i++)
    #pragma unroll
    for (int q = 0; q < 8; q++) acc[i].f[q] = 0.0f;
  #pragma unroll
  for (int i = 0; i < 2; i++)
    acc[i].v = __builtin_amdgcn_wmma_f32_16x16x32_bf16(
        false, a[i].v, false, b.v, (short)0, acc[i].v, false, false);

  #pragma unroll
  for (int i = 0; i < 2; i++)
    #pragma unroll
    for (int j = 0; j < 8; j++) {
      int vch = i * 16 + j + 8 * half;
      O[(size_t)vch * HWDIM + col0 + l15] = f2b(acc[i].f[j]);
    }
}

// ---- host-side orchestration --------------------------------------------

extern "C" void kernel_launch(void* const* d_in, const int* in_sizes, int n_in,
                              void* d_out, int out_size, void* d_ws, size_t ws_size,
                              hipStream_t stream) {
  const float* x  = (const float*)d_in[0];
  const float* Wk = (const float*)d_in[1];
  const float* bk = (const float*)d_in[2];
  const float* Wq = (const float*)d_in[3];
  const float* bq = (const float*)d_in[4];
  const float* Wv = (const float*)d_in[5];
  const float* bv = (const float*)d_in[6];
  const float* Wr = (const float*)d_in[7];
  const float* br = (const float*)d_in[8];
  float* out = (float*)d_out;

  char* ws = (char*)d_ws;
  const size_t SZf = (size_t)NBAT * CDIM * HWDIM * sizeof(float);   // 64 MiB
  const size_t SZb = SZf / 2;                                       // 32 MiB
  float*  Kf   = (float*)(ws);
  float*  Qf   = (float*)(ws + SZf);
  float*  Vf   = (float*)(ws + 2 * SZf);
  __bf16* kb   = (__bf16*)(ws + 3 * SZf);
  __bf16* qb   = (__bf16*)(ws + 3 * SZf + SZb);
  __bf16* vb   = (__bf16*)(ws + 3 * SZf + 2 * SZb);
  __bf16* aggb = (__bf16*)(ws + 3 * SZf + 3 * SZb);
  float*  ctx  = (float*)(ws + 3 * SZf + 4 * SZb);

  dim3 b256(256);
  dim3 gG(HWDIM / 128, CDIM / 128, NBAT);

  gemm256<false, false><<<gG, b256, 0, stream>>>(Wk, bk, x, nullptr, Kf);
  gemm256<false, false><<<gG, b256, 0, stream>>>(Wq, bq, x, nullptr, Qf);
  gemm256<false, false><<<gG, b256, 0, stream>>>(Wv, bv, x, nullptr, Vf);

  softmax_rows<<<dim3(NBAT * CDIM), b256, 0, stream>>>(Kf, kb);
  softmax_chan<<<dim3((NBAT * 8 * HWDIM) / 256), b256, 0, stream>>>(Qf, qb);
  cvt_bf16<<<dim3((NBAT * CDIM * HWDIM) / 1024), b256, 0, stream>>>(Vf, vb);

  ctx_kernel<<<dim3(NBAT * 8), dim3(128), 0, stream>>>(kb, vb, ctx);
  att_kernel<<<dim3(HWDIM / 128, 8, NBAT), b256, 0, stream>>>(ctx, qb, aggb);

  gemm256<true, true><<<gG, b256, 0, stream>>>(Wr, br, aggb, x, out);
}
